// PolysemyBertFlowModel_29755533426755
// MI455X (gfx1250) — matrix operand order repaired and checked by reading, not compiled, not run on hardware
//
#include <hip/hip_runtime.h>
#include <hip/hip_bf16.h>
#include <math.h>

// ---------------------------------------------------------------------------
// PolysemyBertFlow fused pipeline for gfx1250 (MI455X), wave32 + WMMA f16.
// Shapes: B=128, SENTS=2, L=2, S=256, D=1024, DH=64, K=8. SCALE=8.
// ---------------------------------------------------------------------------

typedef __attribute__((ext_vector_type(16))) _Float16 v16h;
typedef __attribute__((ext_vector_type(8)))  _Float16 v8h;
typedef __attribute__((ext_vector_type(8)))  float    v8f;
typedef __attribute__((ext_vector_type(4)))  float    v4f;

#define B_   128
#define S_   256
#define D_   1024
#define DH_  64
#define KW_  8
#define NROW 16      // L*K rows per sentence
#define FEAT 516

union HV { v16h v; v8h h[2]; };

// A-matrix fragment (16x32 f16): lane g*16+m holds K = [g*8..+7, 16+g*8..+7]
__device__ __forceinline__ v16h frag_a(const _Float16* row, int d0, int g) {
  HV u;
  u.h[0] = *(const v8h*)(row + d0 + g * 8);
  u.h[1] = *(const v8h*)(row + d0 + 16 + g * 8);
  return u.v;
}
// B-matrix fragment (32x16 f16): lane g*16+n holds K = [g*8..+7, 16+g*8..+7] of B^T row n
__device__ __forceinline__ v16h frag_b(const _Float16* rowT, int d0, int g) {
  HV u;
  u.h[0] = *(const v8h*)(rowT + d0 + g * 8);
  u.h[1] = *(const v8h*)(rowT + d0 + 16 + g * 8);
  return u.v;
}
__device__ __forceinline__ v8f wmma_f16(v16h a, v16h b, v8f c) {
  return __builtin_amdgcn_wmma_f32_16x16x32_f16(false, a, false, b,
                                                (short)0, c, false, false);
}
__device__ __forceinline__ float gelu_f(float x) {
  return 0.5f * x * (1.0f + erff(x * 0.70710678118654752f));
}

// K=1024 pipelined WMMA dot: A rows from LDS (kernel 1) or global (kernel 2), B^T rows from global.
// Next-B prefetch before each WMMA; the unroller flattens this and the
// scheduler keeps ~4 load-pairs in flight with slack waits (round-2 codegen:
// clause'd global_load_b128 pairs + s_wait_loadcnt with slack, no spills).
__device__ __forceinline__ v8f gemm_k1024(const _Float16* arow,
                                          const _Float16* brow, int g) {
  v8f c = {};
  v16h bcur = frag_b(brow, 0, g);
#pragma unroll 4
  for (int kk = 0; kk < 31; ++kk) {
    v16h bnext = frag_b(brow, (kk + 1) * 32, g);
    c = wmma_f16(frag_a(arow, kk * 32, g), bcur, c);
    bcur = bnext;
  }
  c = wmma_f16(frag_a(arow, 31 * 32, g), bcur, c);
  return c;
}

// ---------------------------------------------------------------------------
// Kernel W: convert the four (64,1024) f32 weight matrices to f16
// ---------------------------------------------------------------------------
__global__ __launch_bounds__(256)
void wconv_kernel(const float* __restrict__ qw, const float* __restrict__ kw,
                  const float* __restrict__ vw, const float* __restrict__ miw,
                  _Float16* __restrict__ qf, _Float16* __restrict__ kf,
                  _Float16* __restrict__ vf, _Float16* __restrict__ mif) {
  int i = blockIdx.x * 256 + threadIdx.x;
  if (i < DH_ * D_) {
    qf[i]  = (_Float16)qw[i];
    kf[i]  = (_Float16)kw[i];
    vf[i]  = (_Float16)vw[i];
    mif[i] = (_Float16)miw[i];
  }
}

// ---------------------------------------------------------------------------
// Kernel 1: per-batch gather + max/avg pooling + Q/K/V (WMMA) + attention
// (WMMA) + e-matrix row reductions -> feature[0..255] and svec (512x1024 f16)
// One block per batch, 256 threads = 8 waves.
// ---------------------------------------------------------------------------
__global__ __launch_bounds__(256)
void gather_qkv_att_kernel(const float* __restrict__ token,
                           const int* __restrict__ kwidx,
                           const _Float16* __restrict__ qwf,
                           const _Float16* __restrict__ kwf,
                           const _Float16* __restrict__ vwf,
                           const float* __restrict__ qb,
                           const float* __restrict__ kb,
                           const float* __restrict__ vb,
                           _Float16* __restrict__ svec,
                           float* __restrict__ feature) {
  __shared__ __align__(16) _Float16 sKW[NROW * 1032];   // gathered rows (f16), padded stride
  __shared__ __align__(16) _Float16 sQ[2][16][72];      // q per sent, row-major
  __shared__ __align__(16) _Float16 sK[2][16][72];      // k per sent, row-major
  __shared__ __align__(16) _Float16 sVT[2][64][40];     // v per sent, TRANSPOSED, cols>=16 zero
  __shared__ __align__(16) _Float16 sATT[2][16][40];    // att as 16x32 A-source, cols>=16 zero
  __shared__ float sRed[2][64][2][2];                   // [e][h][g][{max,sum}]

  const int b    = blockIdx.x;
  const int tid  = threadIdx.x;
  // wave index is uniform across the wave: pin it to an SGPR so all derived
  // control flow (task selection, store-target selection) is scalar branches
  // instead of per-lane EXEC predication.
  const int wave = __builtin_amdgcn_readfirstlane(tid >> 5);
  const int lane = tid & 31;
  const int lm   = lane & 15;
  const int g    = lane >> 4;

  // zero padded regions once (avoid NaN garbage feeding WMMA)
  {
    _Float16* z1 = &sVT[0][0][0];
    for (int i = tid; i < 2 * 64 * 40; i += 256) z1[i] = (_Float16)0.0f;
    _Float16* z2 = &sATT[0][0][0];
    for (int i = tid; i < 2 * 16 * 40; i += 256) z2[i] = (_Float16)0.0f;
  }
  __syncthreads();

  for (int s = 0; s < 2; ++s) {
    // ---- gather 16 keyword rows (idx reversed across sentences) ----
    const int d = tid * 4;  // 256 threads x 4 cols = 1024
    float mx0 = -INFINITY, mx1 = -INFINITY, mx2 = -INFINITY, mx3 = -INFINITY;
    float sm0 = 0.f, sm1 = 0.f, sm2 = 0.f, sm3 = 0.f;
    for (int n = 0; n < NROW; ++n) {
      const int l = n >> 3, j = n & 7;
      const int row = kwidx[(b * 2 + (1 - s)) * KW_ + j];
      // rows are touched exactly once across the whole dispatch -> NT hint
      const v4f t4 = __builtin_nontemporal_load((const v4f*)(token +
          ((((size_t)b * 2 + s) * 2 + l) * S_ + row) * D_ + d));
      mx0 = fmaxf(mx0, t4.x); mx1 = fmaxf(mx1, t4.y);
      mx2 = fmaxf(mx2, t4.z); mx3 = fmaxf(mx3, t4.w);
      sm0 += t4.x; sm1 += t4.y; sm2 += t4.z; sm3 += t4.w;
      _Float16* dst = &sKW[n * 1032 + d];
      dst[0] = (_Float16)t4.x; dst[1] = (_Float16)t4.y;
      dst[2] = (_Float16)t4.z; dst[3] = (_Float16)t4.w;
    }
    // s_max = max over all 16 rows; s_avg = mean over all 16 rows
    {
      const size_t r0 = ((size_t)b * 4 + s * 2) * D_;
      svec[r0 + d + 0] = (_Float16)mx0;
      svec[r0 + d + 1] = (_Float16)mx1;
      svec[r0 + d + 2] = (_Float16)mx2;
      svec[r0 + d + 3] = (_Float16)mx3;
      const float inv16 = 1.0f / 16.0f;
      svec[r0 + D_ + d + 0] = (_Float16)(sm0 * inv16);
      svec[r0 + D_ + d + 1] = (_Float16)(sm1 * inv16);
      svec[r0 + D_ + d + 2] = (_Float16)(sm2 * inv16);
      svec[r0 + D_ + d + 3] = (_Float16)(sm3 * inv16);
    }
    __syncthreads();

    // ---- Q/K/V: 12 (mat, n-tile) jobs over 8 waves; K=1024 in 32 WMMA steps
    for (int task = wave; task < 12; task += 8) {
      const int mat = __builtin_amdgcn_readfirstlane(task >> 2);
      const int n16 = __builtin_amdgcn_readfirstlane(task & 3);
      // ternary chains keep these in the global address space (no FLAT)
      const _Float16* W  = (mat == 0) ? qwf : ((mat == 1) ? kwf : vwf);
      const float* bptr  = (mat == 0) ? qb  : ((mat == 1) ? kb  : vb);
      const _Float16* arow = &sKW[lm * 1032];
      const _Float16* brow = W + (size_t)(n16 * 16 + lm) * D_;
      v8f c = gemm_k1024(arow, brow, g);
      const int   h    = n16 * 16 + lm;   // D layout: n = lane&15
      const float bias = bptr[h];
      if (mat == 0) {
        for (int r = 0; r < 8; ++r) sQ[s][r + 8 * g][h] = (_Float16)(c[r] + bias);
      } else if (mat == 1) {
        for (int r = 0; r < 8; ++r) sK[s][r + 8 * g][h] = (_Float16)(c[r] + bias);
      } else {
        for (int r = 0; r < 8; ++r) sVT[s][h][r + 8 * g] = (_Float16)(c[r] + bias);
      }
    }
    __syncthreads();
  }

  // ---- att_e = q[e] . k[1-e]^T / 8  (16x16), stored as A-source in LDS ----
  if (wave < 2) {
    const int e = wave;         // e=0: q0.k1^T ; e=1: q1.k0^T
    const _Float16* arow = &sQ[e][lm][0];
    const _Float16* brow = &sK[1 - e][lm][0];
    v8f c = {};
    c = wmma_f16(frag_a(arow, 0, g),  frag_b(brow, 0, g),  c);
    c = wmma_f16(frag_a(arow, 32, g), frag_b(brow, 32, g), c);
    for (int r = 0; r < 8; ++r) {
      const int m = r + 8 * g;
      sATT[e][m][lm] = (_Float16)(c[r] * 0.125f);
    }
  }
  __syncthreads();

  // ---- e_mat = att_e . v[1-e]  (16x64); per-column max/sum partials ----
  {
    const int e   = __builtin_amdgcn_readfirstlane(wave >> 2);
    const int n16 = __builtin_amdgcn_readfirstlane(wave & 3);
    const int h = n16 * 16 + lm;
    v8f c = {};
    c = wmma_f16(frag_a(&sATT[e][lm][0], 0, g),
                 frag_b(&sVT[1 - e][h][0], 0, g), c);
    float pmax = -INFINITY, psum = 0.f;
    for (int r = 0; r < 8; ++r) { pmax = fmaxf(pmax, c[r]); psum += c[r]; }
    sRed[e][h][g][0] = pmax;
    sRed[e][h][g][1] = psum;
  }
  __syncthreads();

  // ---- att_feat = [e1max, e1mean, e2max, e2mean] -> feature[0..255] ----
  if (tid < 128) {
    const int e = tid >> 6, h = tid & 63;
    const float mx = fmaxf(sRed[e][h][0][0], sRed[e][h][1][0]);
    const float mn = (sRed[e][h][0][1] + sRed[e][h][1][1]) * (1.0f / 16.0f);
    float* f = feature + (size_t)b * FEAT;
    f[e * 128 + h]      = mx;
    f[e * 128 + 64 + h] = mn;
  }
}

// ---------------------------------------------------------------------------
// Kernel 2: mi projection GEMM (512 x 1024) . (1024 x 64) via WMMA.
// 128 wave-tiles: grid 16 blocks x 8 waves.
// ---------------------------------------------------------------------------
__global__ __launch_bounds__(256)
void miproj_kernel(const _Float16* __restrict__ svec,
                   const _Float16* __restrict__ mif,
                   const float* __restrict__ mib,
                   float* __restrict__ proj) {
  const int wave = __builtin_amdgcn_readfirstlane(threadIdx.x >> 5);
  const int lane = threadIdx.x & 31;
  const int lm = lane & 15, g = lane >> 4;
  const int tile  = blockIdx.x * 8 + wave;   // 0..127
  const int mtile = tile >> 2, n16 = tile & 3;
  const _Float16* arow = svec + (size_t)(mtile * 16 + lm) * D_;
  const _Float16* brow = mif + (size_t)(n16 * 16 + lm) * D_;
  v8f c = gemm_k1024(arow, brow, g);
  const int   h    = n16 * 16 + lm;
  const float bias = mib[h];
  for (int r = 0; r < 8; ++r) {
    const int m = mtile * 16 + r + 8 * g;
    proj[(size_t)m * DH_ + h] = c[r] + bias;
  }
}

// ---------------------------------------------------------------------------
// Kernel 3: inter mat-vecs + 4 cosines; copy projected vectors into feature.
// One block per batch, 64 threads.
// ---------------------------------------------------------------------------
__global__ __launch_bounds__(64)
void inter_cos_kernel(const float* __restrict__ proj,
                      const float* __restrict__ interw,
                      const float* __restrict__ interb,
                      float* __restrict__ feature) {
  __shared__ float sP[4][64];
  __shared__ float sU[4][64];
  __shared__ float sDot[64], sNa[64], sNb[64];
  const int b = blockIdx.x, t = threadIdx.x;
  for (int i = 0; i < 4; ++i) sP[i][t] = proj[((size_t)b * 4 + i) * DH_ + t];
  __syncthreads();
  float* f = feature + (size_t)b * FEAT;
  for (int i = 0; i < 4; ++i) {
    const int xs = (i < 2) ? 0 : 1;                  // s1_max for i0,i1; s1_avg for i2,i3
    float acc = interb[i * 64 + t];
    const float* wr = interw + ((size_t)i * 64 + t) * 64;
    for (int dd = 0; dd < 64; ++dd) acc += sP[xs][dd] * wr[dd];
    sU[i][t] = acc;
    f[260 + i * 64 + t] = sP[i][t];                  // s1_max, s1_avg, s2_max, s2_avg
  }
  __syncthreads();
  for (int i = 0; i < 4; ++i) {
    const int ys = (i & 1) ? 3 : 2;                  // s2_max / s2_avg partner
    const float u = sU[i][t], y = sP[ys][t];
    sDot[t] = u * y; sNa[t] = u * u; sNb[t] = y * y;
    __syncthreads();
    for (int off = 32; off > 0; off >>= 1) {
      if (t < off) { sDot[t] += sDot[t + off]; sNa[t] += sNa[t + off]; sNb[t] += sNb[t + off]; }
      __syncthreads();
    }
    if (t == 0) {
      const float na = fmaxf(sqrtf(sNa[0]), 1e-8f);
      const float nb = fmaxf(sqrtf(sNb[0]), 1e-8f);
      f[256 + i] = sDot[0] / (na * nb);
    }
    __syncthreads();
  }
}

// ---------------------------------------------------------------------------
// Kernel 4: fc0 -> gelu -> fc1 -> gelu -> softmax. One block per batch.
// ---------------------------------------------------------------------------
__global__ __launch_bounds__(64)
void head_kernel(const float* __restrict__ feature,
                 const float* __restrict__ fc0w, const float* __restrict__ fc0b,
                 const float* __restrict__ fc1w, const float* __restrict__ fc1b,
                 float* __restrict__ out) {
  __shared__ float sF[520];
  __shared__ float sH[64];
  __shared__ float sZ[2];
  const int b = blockIdx.x, t = threadIdx.x;
  for (int i = t; i < FEAT; i += 64) sF[i] = feature[(size_t)b * FEAT + i];
  __syncthreads();
  float acc = fc0b[t];
  const float* wr = fc0w + (size_t)t * FEAT;
  for (int j = 0; j < FEAT; ++j) acc += sF[j] * wr[j];
  sH[t] = gelu_f(acc);
  __syncthreads();
  if (t < 2) {
    float z = fc1b[t];
    const float* w1 = fc1w + t * 64;
    for (int j = 0; j < 64; ++j) z += sH[j] * w1[j];
    sZ[t] = gelu_f(z);
  }
  __syncthreads();
  if (t < 2) {
    const float m  = fmaxf(sZ[0], sZ[1]);
    const float e0 = __expf(sZ[0] - m), e1 = __expf(sZ[1] - m);
    const float e  = (t == 0) ? e0 : e1;
    out[(size_t)b * 2 + t] = e / (e0 + e1);
  }
}

// ---------------------------------------------------------------------------
extern "C" void kernel_launch(void* const* d_in, const int* in_sizes, int n_in,
                              void* d_out, int out_size, void* d_ws, size_t ws_size,
                              hipStream_t stream) {
  const float* token  = (const float*)d_in[0];
  const int*   kwidx  = (const int*)d_in[1];
  const float* qw     = (const float*)d_in[2];
  const float* qb     = (const float*)d_in[3];
  const float* kw     = (const float*)d_in[4];
  const float* kb     = (const float*)d_in[5];
  const float* vw     = (const float*)d_in[6];
  const float* vb     = (const float*)d_in[7];
  const float* miw    = (const float*)d_in[8];
  const float* mib    = (const float*)d_in[9];
  const float* interw = (const float*)d_in[10];
  const float* interb = (const float*)d_in[11];
  const float* fc0w   = (const float*)d_in[12];
  const float* fc0b   = (const float*)d_in[13];
  const float* fc1w   = (const float*)d_in[14];
  const float* fc1b   = (const float*)d_in[15];
  float* out = (float*)d_out;

  // workspace carve-out (all offsets 256B-aligned); total ~2.0 MB
  char* ws = (char*)d_ws;
  _Float16* qf      = (_Float16*)(ws);                               // 128 KB
  _Float16* kf      = (_Float16*)(ws + 131072);                      // 128 KB
  _Float16* vf      = (_Float16*)(ws + 262144);                      // 128 KB
  _Float16* mif     = (_Float16*)(ws + 393216);                      // 128 KB
  _Float16* svec    = (_Float16*)(ws + 524288);                      // 1 MB (512 x 1024 f16)
  float*    proj    = (float*)(ws + 524288 + 1048576);               // 128 KB (512 x 64 f32)
  float*    feature = (float*)(ws + 524288 + 1048576 + 131072);      // 258 KB (128 x 516 f32)

  wconv_kernel<<<256, 256, 0, stream>>>(qw, kw, vw, miw, qf, kf, vf, mif);
  gather_qkv_att_kernel<<<B_, 256, 0, stream>>>(token, kwidx, qf, kf, vf,
                                                qb, kb, vb, svec, feature);
  miproj_kernel<<<16, 256, 0, stream>>>(svec, mif, mib, proj);
  inter_cos_kernel<<<B_, 64, 0, stream>>>(proj, interw, interb, feature);
  head_kernel<<<B_, 64, 0, stream>>>(feature, fc0w, fc0b, fc1w, fc1b, out);
}
